// MixtureOfAggregators_72610717106255
// MI455X (gfx1250) — compile-verified, matrix operand order checked by reading, not run on hardware
//
#include <hip/hip_runtime.h>
#include <hip/hip_bf16.h>

typedef unsigned short u16;
typedef __attribute__((ext_vector_type(16))) __bf16 v16bf;
typedef __attribute__((ext_vector_type(8)))  float  v8f;
typedef __attribute__((ext_vector_type(4)))  unsigned u32x4;
typedef __attribute__((ext_vector_type(8)))  unsigned u32x8;

#define B_   16
#define N_   1024
#define D_   2048
#define E_   8
#define H_   512
#define A_   256
#define L_   64
#define C_   8
#define M_   (B_ * N_)          /* 16384 rows of x, flattened (b,n) */

// ---------- bf16 helpers (round-to-nearest-even) ----------
__device__ __forceinline__ u16 f2bf(float f) {
  unsigned u = __float_as_uint(f);
  unsigned r = u + 0x7FFFu + ((u >> 16) & 1u);
  return (u16)(r >> 16);
}
__device__ __forceinline__ float bf2f(u16 h) {
  return __uint_as_float(((unsigned)h) << 16);
}

// ---------- WMMA fragment load ----------
// A 16x32 bf16 fragment (ISA 7.12.2): lane l holds row (l%16); 16 halfs as two
// contiguous 8-half chunks at k-offsets {0,16} + (l>=16 ? 8 : 0).
// B 32x16 mirrors with lane = column (weights pre-transposed).
__device__ __forceinline__ v16bf load_frag(const u16* __restrict__ base, int stride, int lane) {
  const int r    = lane & 15;
  const int koff = (lane & 16) ? 8 : 0;
  const u16* p = base + r * stride + koff;
  union { v16bf v; uint4 q[2]; } u;
  u.q[0] = *(const uint4*)(p);
  u.q[1] = *(const uint4*)(p + 16);
  return u.v;
}

__device__ __forceinline__ v8f wmma_bf16(v16bf a, v16bf b, v8f c) {
  return __builtin_amdgcn_wmma_f32_16x16x32_bf16(false, a, false, b, (short)0, c, false, false);
}

// ---------- CDNA5 async data movement ----------
// Per-lane 16B global -> LDS, tracked with ASYNCcnt.
__device__ __forceinline__ void async_cp16(unsigned lds_addr, const void* gptr) {
  asm volatile("global_load_async_to_lds_b128 %0, %1, off"
               :: "v"(lds_addr), "v"((unsigned long long)(uintptr_t)gptr)
               : "memory");
}
__device__ __forceinline__ void wait_async_le2() {
  asm volatile("s_wait_asynccnt 0x2" ::: "memory");
}
__device__ __forceinline__ void wait_async_le0() {
  asm volatile("s_wait_asynccnt 0x0" ::: "memory");
}

// Tensor Data Mover: 2D tile (tile_dim0=32 halfs x tile_dim1 rows) from a
// row-major bf16 matrix with row stride D_ into LDS, padding 16B after every
// 64B row (pad_interval=3 -> 16 DWORDs, pad_amount=3 -> 4 DWORDs) to give the
// LDSP=40-half padded row stride the WMMA fragment loader expects.
__device__ __forceinline__ void tdm_load_tile(unsigned lds_addr, const void* gptr,
                                              unsigned rows) {
  unsigned long long ga = (unsigned long long)(uintptr_t)gptr;
  u32x4 g0;
  g0[0] = 1u;                                  // count=1, user mode
  g0[1] = lds_addr;                            // lds_addr (bytes)
  g0[2] = (unsigned)ga;                        // global_addr[31:0]
  g0[3] = (unsigned)(ga >> 32) | (2u << 30);   // global_addr[56:32] | type=2
  u32x8 g1;
  g1[0] = (1u << 16) | (1u << 20) | (3u << 22) | (3u << 25); // data_size=2B, pad_en, 16dw interval, 4dw pad
  g1[1] = ((unsigned)D_ & 0xFFFFu) << 16;      // tensor_dim0 = 2048 (lo16)
  g1[2] = ((unsigned)H_ & 0xFFFFu) << 16;      // dim0 hi=0 | tensor_dim1 = 512 (lo16)
  g1[3] = 32u << 16;                           // dim1 hi=0 | tile_dim0 = 32
  g1[4] = rows;                                // tile_dim1 = rows | tile_dim2 = 0
  g1[5] = (unsigned)D_;                        // tensor_dim0_stride lo32 = 2048
  g1[6] = 0u;                                  // stride hi | dim1_stride lo
  g1[7] = 0u;
  asm volatile("tensor_load_to_lds %0, %1" :: "s"(g0), "s"(g1) : "memory");
}

// ---------- conversion / layout kernels ----------
__global__ void k_f32_to_bf16(const float* __restrict__ src, u16* __restrict__ dst, int n) {
  for (int i = blockIdx.x * blockDim.x + threadIdx.x; i < n; i += gridDim.x * blockDim.x)
    dst[i] = f2bf(src[i]);
}

// W1T[e][n][k] (e<8 from eW1[e][k][n], e==8 from router_W[k][n]); 9*512*2048 elems
__global__ void k_build_w1t(const float* __restrict__ eW1, const float* __restrict__ rW,
                            u16* __restrict__ dst, int n) {
  for (int i = blockIdx.x * blockDim.x + threadIdx.x; i < n; i += gridDim.x * blockDim.x) {
    int e   = i / (H_ * D_);
    int rem = i - e * (H_ * D_);
    int nn  = rem >> 11;          // /2048
    int k   = rem & (D_ - 1);
    float v = (e < E_) ? eW1[(e * D_ + k) * H_ + nn] : rW[k * H_ + nn];
    dst[i] = f2bf(v);
  }
}

// src [E][H][A] -> dst [E][A][H]  (for eWv / eWu)
__global__ void k_transpose_ha(const float* __restrict__ src, u16* __restrict__ dst, int n) {
  for (int i = blockIdx.x * blockDim.x + threadIdx.x; i < n; i += gridDim.x * blockDim.x) {
    int e   = i / (A_ * H_);
    int rem = i - e * (A_ * H_);
    int a   = rem >> 9;           // /512
    int h   = rem & (H_ - 1);
    dst[i] = f2bf(src[(e * H_ + h) * A_ + a]);
  }
}

// ---------- GEMM1: H = relu(Xbf[16384,2048] @ W1T_e^T[2048,512] + b1) -> bf16 ----------
// block tile 128x128, 8 waves (4x2), wave tile 32x64 = 2x4 WMMA tiles, K-step 32.
// Double-buffered LDS; A tile via GLOBAL_LOAD_ASYNC_TO_LDS, B tile via TDM.
#define G1_BM 128
#define G1_BN 128
#define G1_BK 32
#define LDSP  40   /* padded LDS row stride in halfs (80B = 64B row + 16B TDM pad) */

__device__ __forceinline__ void g1_issue_tileA(const u16* __restrict__ Xbf, int bm, int kb,
                                               unsigned as_base, int t) {
  const int r = t >> 2, c = t & 3;   // 128 rows x 4 chunks; 2 chunks/thread
  async_cp16(as_base + (unsigned)(r * LDSP + c * 8) * 2,
             &Xbf[(bm + r) * D_ + kb + c * 8]);
  async_cp16(as_base + (unsigned)((r + 64) * LDSP + c * 8) * 2,
             &Xbf[(bm + r + 64) * D_ + kb + c * 8]);
}

__global__ __launch_bounds__(256) void k_gemm1(
    const u16* __restrict__ Xbf,    // [M_][D_]
    const u16* __restrict__ W1T,    // [H_][D_] for this expert
    const float* __restrict__ bias, // [H_]
    u16* __restrict__ Hout)         // [M_][H_]
{
  __shared__ u16 As[2][G1_BM * LDSP];
  __shared__ u16 Bs[2][G1_BN * LDSP];

  const int t    = threadIdx.x;
  const int lane = t & 31;
  const int wid  = t >> 5;     // 0..7
  const int wm   = wid & 3;    // 0..3 -> 32-row slice
  const int wn   = wid >> 2;   // 0..1 -> 64-col slice
  const int bm   = blockIdx.x * G1_BM;
  const int bn   = blockIdx.y * G1_BN;

  const unsigned asb0 = (unsigned)(uintptr_t)&As[0][0];
  const unsigned asb1 = (unsigned)(uintptr_t)&As[1][0];
  const unsigned bsb0 = (unsigned)(uintptr_t)&Bs[0][0];
  const unsigned bsb1 = (unsigned)(uintptr_t)&Bs[1][0];

  v8f acc[2][4] = {};

  // prologue: stage tile 0 into buffer 0
  g1_issue_tileA(Xbf, bm, 0, asb0, t);
  if (wid == 0) tdm_load_tile(bsb0, &W1T[bn * D_], G1_BN);

  int buf = 0;
  for (int kb = 0; kb < D_; kb += G1_BK) {
    const bool last = (kb + G1_BK >= D_);
    if (!last) {
      // stage tile k+1 into the other buffer (its readers finished last iter)
      g1_issue_tileA(Xbf, bm, kb + G1_BK, buf ? asb0 : asb1, t);
      if (wid == 0) tdm_load_tile(buf ? bsb0 : bsb1, &W1T[bn * D_ + kb + G1_BK], G1_BN);
      wait_async_le2();                               // tile k done; k+1 in flight
      if (wid == 0) __builtin_amdgcn_s_wait_tensorcnt(1);
    } else {
      wait_async_le0();
      if (wid == 0) __builtin_amdgcn_s_wait_tensorcnt(0);
    }
    __syncthreads();

    const u16* ab = &As[buf][0];
    const u16* bb = &Bs[buf][0];
    v16bf a0 = load_frag(ab + (wm * 32 + 0)  * LDSP, LDSP, lane);
    v16bf a1 = load_frag(ab + (wm * 32 + 16) * LDSP, LDSP, lane);
    v16bf b0 = load_frag(bb + (wn * 64 + 0)  * LDSP, LDSP, lane);
    v16bf b1 = load_frag(bb + (wn * 64 + 16) * LDSP, LDSP, lane);
    v16bf b2 = load_frag(bb + (wn * 64 + 32) * LDSP, LDSP, lane);
    v16bf b3 = load_frag(bb + (wn * 64 + 48) * LDSP, LDSP, lane);

    acc[0][0] = wmma_bf16(a0, b0, acc[0][0]);
    acc[0][1] = wmma_bf16(a0, b1, acc[0][1]);
    acc[0][2] = wmma_bf16(a0, b2, acc[0][2]);
    acc[0][3] = wmma_bf16(a0, b3, acc[0][3]);
    acc[1][0] = wmma_bf16(a1, b0, acc[1][0]);
    acc[1][1] = wmma_bf16(a1, b1, acc[1][1]);
    acc[1][2] = wmma_bf16(a1, b2, acc[1][2]);
    acc[1][3] = wmma_bf16(a1, b3, acc[1][3]);
    __syncthreads();
    buf ^= 1;
  }

  // epilogue: bias + relu, store bf16.  C layout: M = r + 8*(lane>=16), N = lane%16
  const int ml = lane & 15;
  const int mh = (lane >> 4) * 8;
  for (int tm = 0; tm < 2; ++tm)
    for (int tn = 0; tn < 4; ++tn) {
      int col  = bn + wn * 64 + tn * 16 + ml;
      float bv = bias[col];
      for (int r = 0; r < 8; ++r) {
        int row = bm + wm * 32 + tm * 16 + mh + r;
        float v = acc[tm][tn][r] + bv;
        v = v > 0.f ? v : 0.f;
        Hout[row * H_ + col] = f2bf(v);
      }
    }
}

// ---------- scores: Av/Au WMMA + gated reduction over A=256 ----------
// block = 128 rows (8 waves x 16 rows); frags read straight from global (L2-resident)
__global__ __launch_bounds__(256) void k_scores(
    const u16* __restrict__ Hbf,   // [M_][H_]
    const u16* __restrict__ WvT,   // [A_][H_]
    const u16* __restrict__ WuT,   // [A_][H_]
    const float* __restrict__ ebv, const float* __restrict__ ebu,
    const float* __restrict__ ew,  const float* __restrict__ ebw,
    float* __restrict__ scores)    // [M_]
{
  __shared__ float s_sc[128];
  const int t    = threadIdx.x;
  const int lane = t & 31;
  const int wid  = t >> 5;
  const int m0   = blockIdx.x * 128 + wid * 16;

  for (int i = t; i < 128; i += 256) s_sc[i] = 0.f;
  __syncthreads();

  const int ml = lane & 15;
  const int mh = (lane >> 4) * 8;
  float ps[8] = {};

  for (int at = 0; at < A_ / 16; ++at) {
    v8f av = {}, au = {};
    const int a0 = at * 16;
    for (int kb = 0; kb < H_; kb += 32) {
      v16bf af = load_frag(&Hbf[m0 * H_ + kb], H_, lane);
      v16bf bv = load_frag(&WvT[a0 * H_ + kb], H_, lane);
      v16bf bu = load_frag(&WuT[a0 * H_ + kb], H_, lane);
      av = wmma_bf16(af, bv, av);
      au = wmma_bf16(af, bu, au);
    }
    const int a = a0 + ml;
    const float bva = ebv[a], bua = ebu[a], wa = ew[a];
    for (int r = 0; r < 8; ++r) {
      float tv = tanhf(av[r] + bva);
      float su = 1.f / (1.f + __expf(-(au[r] + bua)));
      ps[r] += tv * su * wa;
    }
  }
  for (int r = 0; r < 8; ++r)
    atomicAdd(&s_sc[wid * 16 + mh + r], ps[r]);
  __syncthreads();

  const float bw = ebw[0];
  for (int i = t; i < 128; i += 256)
    scores[blockIdx.x * 128 + i] = s_sc[i] + bw;
}

// ---------- per-bag softmax over N=1024 instances ----------
__global__ __launch_bounds__(256) void k_softmax(const float* __restrict__ s,
                                                 float* __restrict__ attn) {
  __shared__ float red[256];
  const int b = blockIdx.x, t = threadIdx.x;
  const float* sb = s + b * N_;

  float mx = -1e30f;
  for (int i = t; i < N_; i += 256) mx = fmaxf(mx, sb[i]);
  red[t] = mx; __syncthreads();
  for (int w = 128; w > 0; w >>= 1) { if (t < w) red[t] = fmaxf(red[t], red[t + w]); __syncthreads(); }
  mx = red[0]; __syncthreads();

  float sum = 0.f;
  for (int i = t; i < N_; i += 256) sum += __expf(sb[i] - mx);
  red[t] = sum; __syncthreads();
  for (int w = 128; w > 0; w >>= 1) { if (t < w) red[t] += red[t + w]; __syncthreads(); }
  const float inv = 1.f / red[0];
  for (int i = t; i < N_; i += 256) attn[b * N_ + i] = __expf(sb[i] - mx) * inv;
}

// ---------- pooled[b][h] = sum_n w_n * H[b,n,h]  (attn==nullptr -> mean) ----------
__global__ __launch_bounds__(256) void k_pooled(const u16* __restrict__ Hbf,
                                                const float* __restrict__ attn,
                                                float* __restrict__ out) { // [B_][H_]
  const int b = blockIdx.x, t = threadIdx.x;
  const int base = b * N_;
  float a0 = 0.f, a1 = 0.f;
  for (int n = 0; n < N_; ++n) {
    float w = attn ? attn[base + n] : (1.0f / (float)N_);
    const u16* hp = &Hbf[(base + n) * H_];
    if (n + 8 < N_) __builtin_prefetch(&Hbf[(base + n + 8) * H_ + t], 0, 3);
    a0 += w * bf2f(hp[t]);
    a1 += w * bf2f(hp[t + 256]);
  }
  out[b * H_ + t]       = a0;
  out[b * H_ + t + 256] = a1;
}

// ---------- tail: router softmax + top2 + latent_e/logit_e + sparse combine ----------
__global__ __launch_bounds__(256) void k_tail(
    const float* __restrict__ Rbuf,  // [B_][H_]  router mean features
    const float* __restrict__ Pool,  // [E_][B_][H_]
    const float* __restrict__ fc_W, const float* __restrict__ fc_b,
    const float* __restrict__ eW2,  const float* __restrict__ eb2,
    const float* __restrict__ eWc,  const float* __restrict__ ebc,
    float* __restrict__ out)         // latent @0 (16*64), logits @1024 (16*8), g_soft @1152
{
  __shared__ float lg[E_];
  __shared__ float comb[E_];
  __shared__ float lat[E_ * L_];
  __shared__ float lge[E_ * C_];
  const int b = blockIdx.x, t = threadIdx.x;

  if (t < E_) lg[t] = fc_b[t];
  __syncthreads();
  {
    float part[E_] = {};
    for (int h = t; h < H_; h += 256) {
      float xr = Rbuf[b * H_ + h];
      for (int e = 0; e < E_; ++e) part[e] += xr * fc_W[h * E_ + e];
    }
    for (int e = 0; e < E_; ++e) atomicAdd(&lg[e], part[e]);
  }
  __syncthreads();

  if (t == 0) {
    float mx = lg[0];
    for (int e = 1; e < E_; ++e) mx = fmaxf(mx, lg[e]);
    float gs[E_], s = 0.f;
    for (int e = 0; e < E_; ++e) { gs[e] = __expf(lg[e] - mx); s += gs[e]; }
    float inv = 1.f / s;
    int i1 = 0; float v1 = -1.f;
    for (int e = 0; e < E_; ++e) {
      gs[e] *= inv;
      out[1152 + b * E_ + e] = gs[e];            // g_soft
      if (gs[e] > v1) { v1 = gs[e]; i1 = e; }
    }
    int i2 = 0; float v2 = -1.f;
    for (int e = 0; e < E_; ++e) if (e != i1 && gs[e] > v2) { v2 = gs[e]; i2 = e; }
    float denom = v1 + v2 + 1e-8f;
    for (int e = 0; e < E_; ++e) comb[e] = 0.f;
    comb[i1] = v1 / denom;
    comb[i2] = v2 / denom;
  }
  __syncthreads();

  // latent_e = relu(Pool[e,b,:] @ eW2[e] + eb2[e])
  for (int i = t; i < E_ * L_; i += 256) {
    int e = i >> 6, j = i & (L_ - 1);
    const float* pp = &Pool[(e * B_ + b) * H_];
    const float* wp = &eW2[(size_t)e * H_ * L_ + j];
    float s = eb2[e * L_ + j];
    for (int h = 0; h < H_; ++h) s += pp[h] * wp[h * L_];
    lat[i] = s > 0.f ? s : 0.f;
  }
  __syncthreads();

  if (t < E_ * C_) {
    int e = t >> 3, c = t & (C_ - 1);
    float s = ebc[e * C_ + c];
    for (int l = 0; l < L_; ++l) s += lat[e * L_ + l] * eWc[(e * L_ + l) * C_ + c];
    lge[t] = s;
  }
  __syncthreads();

  if (t < L_) {
    float s = 0.f;
    for (int e = 0; e < E_; ++e) s += comb[e] * lat[e * L_ + t];
    out[b * L_ + t] = s;                         // latent
  } else if (t < L_ + C_) {
    int c = t - L_;
    float s = 0.f;
    for (int e = 0; e < E_; ++e) s += comb[e] * lge[e * C_ + c];
    out[1024 + b * C_ + c] = s;                  // logits
  }
}

// ---------------------------------------------------------------------------
extern "C" void kernel_launch(void* const* d_in, const int* in_sizes, int n_in,
                              void* d_out, int out_size, void* d_ws, size_t ws_size,
                              hipStream_t stream) {
  (void)in_sizes; (void)n_in; (void)out_size; (void)ws_size;
  const float* x        = (const float*)d_in[0];
  const float* router_W = (const float*)d_in[1];
  const float* router_b = (const float*)d_in[2];
  const float* fc_W     = (const float*)d_in[3];
  const float* fc_b     = (const float*)d_in[4];
  const float* eW1      = (const float*)d_in[5];
  const float* eb1      = (const float*)d_in[6];
  const float* eWv      = (const float*)d_in[7];
  const float* ebv      = (const float*)d_in[8];
  const float* eWu      = (const float*)d_in[9];
  const float* ebu      = (const float*)d_in[10];
  const float* ew       = (const float*)d_in[11];
  const float* ebw      = (const float*)d_in[12];
  const float* eW2      = (const float*)d_in[13];
  const float* eb2      = (const float*)d_in[14];
  const float* eWc      = (const float*)d_in[15];
  const float* ebc      = (const float*)d_in[16];
  float* out = (float*)d_out;

  char* base = (char*)d_ws;
  size_t off = 0;
  auto take = [&](size_t bytes) -> void* {
    off = (off + 255) & ~(size_t)255;
    void* r = base + off; off += bytes; return r;
  };

  u16*   Xbf    = (u16*)take((size_t)M_ * D_ * 2);            // 67 MB
  u16*   W1T    = (u16*)take((size_t)(E_ + 1) * H_ * D_ * 2); // 19 MB
  u16*   WvT    = (u16*)take((size_t)E_ * A_ * H_ * 2);       // 2 MB
  u16*   WuT    = (u16*)take((size_t)E_ * A_ * H_ * 2);       // 2 MB
  u16*   Hbf    = (u16*)take((size_t)M_ * H_ * 2);            // 17 MB (reused per expert)
  float* scores = (float*)take((size_t)M_ * 4);
  float* attn   = (float*)take((size_t)M_ * 4);
  float* Pool   = (float*)take((size_t)E_ * B_ * H_ * 4);
  float* Rbuf   = (float*)take((size_t)B_ * H_ * 4);

  // ---- conversions / layout ----
  k_f32_to_bf16<<<4096, 256, 0, stream>>>(x, Xbf, M_ * D_);
  k_build_w1t<<<4096, 256, 0, stream>>>(eW1, router_W, W1T, (E_ + 1) * H_ * D_);
  k_transpose_ha<<<1024, 256, 0, stream>>>(eWv, WvT, E_ * A_ * H_);
  k_transpose_ha<<<1024, 256, 0, stream>>>(eWu, WuT, E_ * A_ * H_);

  // ---- per-expert pipeline (expert E_ == router projection) ----
  for (int e = 0; e <= E_; ++e) {
    const u16*   w1e = W1T + (size_t)e * H_ * D_;
    const float* b1  = (e < E_) ? (eb1 + e * H_) : router_b;
    k_gemm1<<<dim3(M_ / G1_BM, H_ / G1_BN), 256, 0, stream>>>(Xbf, w1e, b1, Hbf);

    if (e < E_) {
      k_scores<<<M_ / 128, 256, 0, stream>>>(Hbf,
          WvT + (size_t)e * A_ * H_, WuT + (size_t)e * A_ * H_,
          ebv + e * A_, ebu + e * A_, ew + e * A_, ebw + e, scores);
      k_softmax<<<B_, 256, 0, stream>>>(scores, attn);
      k_pooled<<<B_, 256, 0, stream>>>(Hbf, attn, Pool + (size_t)e * B_ * H_);
    } else {
      k_pooled<<<B_, 256, 0, stream>>>(Hbf, nullptr, Rbuf);   // router mean
    }
  }

  // ---- tail ----
  k_tail<<<B_, 256, 0, stream>>>(Rbuf, Pool, fc_W, fc_b, eW2, eb2, eWc, ebc, out);
}
